// DynamicMultiRNN_30296699306113
// MI455X (gfx1250) — matrix operand order repaired and checked by reading, not compiled, not run on hardware
//
#include <hip/hip_runtime.h>

// ---------------- problem constants (fixed by the reference) ----------------
#define TT 200
#define BB 256
#define DD 512
#define KK 1024           // 2*D  (GEMM K)
#define ND 2048           // 4*D  (GEMM N, gate-major: i|j|f|o)
#define XST (TT*DD)       // row stride of rnn_input / output (102400)
#define NSTATE (BB*DD)    // 131072 floats per state buffer
#define NWG 512           // 16 M-tiles * 32 D-tiles

typedef __attribute__((ext_vector_type(16))) __bf16 v16bf;
typedef __attribute__((ext_vector_type(8)))  float  v8f;
typedef __attribute__((ext_vector_type(4)))  int    v4i;

union Frag16 { v16bf v; uint4 q[2]; };

// ------------- CDNA5 async global->LDS (guarded; sync fallback) -------------
#if defined(__gfx1250__) && __has_builtin(__builtin_amdgcn_global_load_async_to_lds_b128) && __has_builtin(__builtin_amdgcn_s_wait_asynccnt)
#define HAVE_ASYNC_LDS 1
#endif

__device__ __forceinline__ void g2l_b128(void* lds, const void* g) {
#ifdef HAVE_ASYNC_LDS
  // prototype (from hipcc diagnostic): arg0 = <4 x i32> addrspace(1)*,
  // arg1 = LDS side, then imm offset + imm cpol.
  __builtin_amdgcn_global_load_async_to_lds_b128(
      (__attribute__((address_space(1))) v4i*)(unsigned long long)g,
      (__attribute__((address_space(3))) v4i*)(unsigned)(unsigned long long)lds,
      0, 0);
#else
  *(uint4*)lds = *(const uint4*)g;
#endif
}

__device__ __forceinline__ void wait_async0() {
#ifdef HAVE_ASYNC_LDS
  __builtin_amdgcn_s_wait_asynccnt(0);
#endif
}

__device__ __forceinline__ float sigf(float x) { return 1.0f / (1.0f + __expf(-x)); }

// ------------------- device-wide barrier (persistent kernel) ----------------
__device__ void grid_sync(unsigned* cnt, unsigned* gen) {
  __threadfence();
  __syncthreads();
  if (threadIdx.x == 0) {
    unsigned g = __hip_atomic_load(gen, __ATOMIC_RELAXED, __HIP_MEMORY_SCOPE_AGENT);
    unsigned a = __hip_atomic_fetch_add(cnt, 1u, __ATOMIC_ACQ_REL, __HIP_MEMORY_SCOPE_AGENT);
    if (a == (unsigned)gridDim.x - 1u) {
      __hip_atomic_store(cnt, 0u, __ATOMIC_RELAXED, __HIP_MEMORY_SCOPE_AGENT);
      __hip_atomic_store(gen, g + 1u, __ATOMIC_RELEASE, __HIP_MEMORY_SCOPE_AGENT);
    } else {
      while (__hip_atomic_load(gen, __ATOMIC_ACQUIRE, __HIP_MEMORY_SCOPE_AGENT) == g)
        __builtin_amdgcn_s_sleep(1);
    }
  }
  __syncthreads();
}

// -------- one LSTM layer step for tile (mt,dt): GEMM(+bias) + cell ----------
// z = [x,h] @ W + b ; wave g owns gate g. Weights Wt are bf16, N-major
// (Wt[n][k]) so B fragments are contiguous-K ds_load_b128s.
__device__ __forceinline__ void lstm_layer(
    int t, int mt, int dt,
    const float* xbase, int xstride,       // x rows (layer0: rnn_input; layer1: h0n)
    const float* hprev, float* hnext,      // double-buffered h state
    float* cstate,                         // c state (owner-exclusive)
    const __bf16* __restrict__ Wt, const float* __restrict__ bias,
    const int* __restrict__ seqlen,
    float* h_unm,                          // layer0: unmasked h out (layer1 input)
    const float* resid, float* yout, int ystride) // layer1: residual + output
{
  // LDS: A 16x1024 bf16 (row stride 1048 el -> conflict-free b128 reads),
  //      B double-buffered 64 cols x 64 K (col stride 72 el), z-buffer 4x16x17.
  __shared__ __attribute__((aligned(16))) __bf16 ldsA[16 * 1048];
  __shared__ __attribute__((aligned(16))) __bf16 ldsB[2][64 * 72];
  __shared__ float zbuf[4][16][17];

  const int tid  = threadIdx.x;
  const int lane = tid & 31;
  const int wid  = tid >> 5;      // gate: 0=i 1=j 2=f 3=o
  const int colN = lane & 15;     // WMMA N column of this lane
  const int hi   = lane >> 4;     // lane half (K sub-block select)

  // ---- stage A = bf16(concat[x_rows, h_rows]) : 16 rows x 1024 K ----
  {
    const int r = tid >> 3, sub = tid & 7;     // 8 threads per row
    const int b = mt * 16 + r;
    const float* src = (sub < 4) ? (xbase + (size_t)b * xstride + sub * 128)
                                 : (hprev + b * DD + (sub - 4) * 128);
    __bf16* dst = ldsA + r * 1048 + sub * 128;
#pragma unroll
    for (int jj = 0; jj < 32; ++jj) {
      float4 v = ((const float4*)src)[jj];
      union { __bf16 e[4]; uint2 u; } p;
      p.e[0] = (__bf16)v.x; p.e[1] = (__bf16)v.y;
      p.e[2] = (__bf16)v.z; p.e[3] = (__bf16)v.w;
      *(uint2*)(dst + jj * 4) = p.u;
    }
  }

  // bias folded into accumulator init: every M row of this lane's column
  const float bv = bias[wid * 512 + dt * 16 + colN];
  v8f acc = {bv, bv, bv, bv, bv, bv, bv, bv};

  // stage a 64-K chunk of B (64 cols = 4 gates x 16) for this WG's dt tile
  auto stageB = [&](int kc, int buf) {
#pragma unroll
    for (int j = 0; j < 4; ++j) {
      int i = j * 128 + tid;                // 512 x 16B transfers
      int col = i >> 3, part = i & 7;
      int gcol = (col >> 4) * 512 + dt * 16 + (col & 15);
      const __bf16* src = Wt + (size_t)gcol * KK + kc * 64 + part * 8;
      __bf16* dst = &ldsB[buf][col * 72 + part * 8];
      g2l_b128(dst, src);
    }
  };

  stageB(0, 0);
  // A fragment base: lane holds row M = lane&15; elements are K kb..kb+7 and
  // kb+16..kb+23 with kb = hi*8  (ISA 16-bit A 16x32 layout).
  const __bf16* Abase = ldsA + (lane & 15) * 1048 + hi * 8;

  for (int kc = 0; kc < 16; ++kc) {          // 16 chunks x 64 K = 1024
    const int buf = kc & 1;
    wait_async0();
    __syncthreads();                          // chunk 'buf' (and ldsA) ready
    if (kc + 1 < 16) stageB(kc + 1, buf ^ 1); // overlap next chunk
    // B fragment base: lane holds col N, K kb2..kb2+7 and kb2+16..kb2+23, kb2 = hi*8
    const __bf16* Bb = &ldsB[buf][(wid * 16 + colN) * 72 + hi * 8];
#pragma unroll
    for (int ks = 0; ks < 2; ++ks) {          // 2 x K=32 WMMA per chunk
      Frag16 a, b;
      const __bf16* Ak = Abase + kc * 64 + ks * 32;
      a.q[0] = *(const uint4*)(Ak);
      a.q[1] = *(const uint4*)(Ak + 16);
      b.q[0] = *(const uint4*)(Bb + ks * 32);
      b.q[1] = *(const uint4*)(Bb + ks * 32 + 16);
      acc = __builtin_amdgcn_wmma_f32_16x16x32_bf16(
          false, a.v, false, b.v, (short)0, acc, false, false);
    }
    __syncthreads();                          // done reading 'buf'
  }

  // ---- accumulators -> LDS (C/D layout: c[r] = z[r + 8*hi][colN]) ----
#pragma unroll
  for (int r = 0; r < 8; ++r) zbuf[wid][r + 8 * hi][colN] = acc[r];
  __syncthreads();

  // ---- fused LSTM cell + dynamic_rnn masking (2 elements per thread) ----
#pragma unroll
  for (int u = 0; u < 2; ++u) {
    const int e = tid * 2 + u;
    const int row = e >> 4, col = e & 15;
    const int b = mt * 16 + row;
    const int d = dt * 16 + col;
    const int idx = b * DD + d;
    const float zi = zbuf[0][row][col], zj = zbuf[1][row][col];
    const float zf = zbuf[2][row][col], zo = zbuf[3][row][col];
    const float c_old = cstate[idx];
    const float c_new = c_old * sigf(zf + 1.0f) + sigf(zi) * tanhf(zj);
    const float h_new = tanhf(c_new) * sigf(zo);
    const bool valid = t < seqlen[b];
    const float h_old = hprev[idx];
    hnext[idx]  = valid ? h_new : h_old;      // state copy-through
    cstate[idx] = valid ? c_new : c_old;
    if (h_unm) h_unm[idx] = h_new;            // layer1 consumes UNMASKED h
    if (yout)  yout[(size_t)b * ystride + d] = valid ? (h_new + resid[idx]) : 0.0f;
  }
  __syncthreads();
}

// --------------------------- persistent scan kernel -------------------------
__global__ void __launch_bounds__(128)
lstm_persistent(const float* __restrict__ xin, const int* __restrict__ seqlen,
                const float* __restrict__ b0, const float* __restrict__ b1,
                const __bf16* __restrict__ Wt0, const __bf16* __restrict__ Wt1,
                float* __restrict__ out, float* states, unsigned* sync)
{
  float* h0  = states;                 // [2][NSTATE] parity double-buffer
  float* h1  = states + 2 * NSTATE;    // [2][NSTATE]
  float* c0  = states + 4 * NSTATE;
  float* c1  = states + 5 * NSTATE;
  float* h0n = states + 6 * NSTATE;    // unmasked layer-0 h (layer-1 input)
  unsigned* cnt = sync;
  unsigned* gen = sync + 8;

  const int mt = blockIdx.x >> 5;      // 16 M tiles
  const int dt = blockIdx.x & 31;      // 32 D tiles

  for (int t = 0; t < TT; ++t) {
    const int p = t & 1;
    if (t + 1 < TT) {                  // warm L2/WGP$ for next step's x rows
      const int r = threadIdx.x & 15;
      __builtin_prefetch(xin + (size_t)(mt * 16 + r) * XST + (size_t)(t + 1) * DD, 0, 1);
    }
    // layer 0: A = [x_t, h0]
    lstm_layer(t, mt, dt, xin + (size_t)t * DD, XST,
               h0 + p * NSTATE, h0 + (p ^ 1) * NSTATE, c0,
               Wt0, b0, seqlen, h0n, nullptr, nullptr, 0);
    grid_sync(cnt, gen);
    // layer 1 (residual): A = [h0n, h1]
    lstm_layer(t, mt, dt, h0n, DD,
               h1 + p * NSTATE, h1 + (p ^ 1) * NSTATE, c1,
               Wt1, b1, seqlen, nullptr, h0n, out + (size_t)t * DD, XST);
    grid_sync(cnt, gen);
  }
}

// ------------------ prologue: state init + weight transpose -----------------
__global__ void init_ws_kernel(float* st, unsigned* sync, int n) {
  for (int i = blockIdx.x * blockDim.x + threadIdx.x; i < n;
       i += gridDim.x * blockDim.x)
    st[i] = 0.0f;
  if (blockIdx.x == 0 && threadIdx.x < 16) sync[threadIdx.x] = 0u;
}

// W [1024][2048] f32 (K-major) -> Wt [2048][1024] bf16 (N-major), LDS-tiled
__global__ void wconv_kernel(const float* __restrict__ W, __bf16* __restrict__ Wt) {
  __shared__ float tile[32][33];
  const int n = blockIdx.x * 32;
  const int k = blockIdx.y * 32;
  for (int j = threadIdx.y; j < 32; j += 8)
    tile[j][threadIdx.x] = W[(size_t)(k + j) * ND + n + threadIdx.x];
  __syncthreads();
  for (int j = threadIdx.y; j < 32; j += 8)
    Wt[(size_t)(n + j) * KK + k + threadIdx.x] = (__bf16)tile[threadIdx.x][j];
}

// --------------------------------- launcher ---------------------------------
extern "C" void kernel_launch(void* const* d_in, const int* in_sizes, int n_in,
                              void* d_out, int out_size, void* d_ws, size_t ws_size,
                              hipStream_t stream) {
  const float* x      = (const float*)d_in[0];   // [256,200,512]
  const int*   seqlen = (const int*)d_in[1];     // [256]
  const float* W0     = (const float*)d_in[2];   // [1024,2048]
  const float* b0     = (const float*)d_in[3];   // [2048]
  const float* W1     = (const float*)d_in[4];
  const float* b1     = (const float*)d_in[5];
  float* out = (float*)d_out;

  char* ws = (char*)d_ws;
  __bf16* Wt0    = (__bf16*)(ws);                   // 4 MiB
  __bf16* Wt1    = (__bf16*)(ws + (4u << 20));      // 4 MiB
  float*  states = (float*)(ws + (8u << 20));       // 7 * 512 KiB
  unsigned* sync = (unsigned*)(ws + (12u << 20));   // barrier counter + gen

  init_ws_kernel<<<512, 256, 0, stream>>>(states, sync, 7 * NSTATE);
  dim3 tb(32, 8), tg(ND / 32, KK / 32);
  wconv_kernel<<<tg, tb, 0, stream>>>(W0, Wt0);
  wconv_kernel<<<tg, tb, 0, stream>>>(W1, Wt1);
  lstm_persistent<<<NWG, 128, 0, stream>>>(x, seqlen, b0, b1, Wt0, Wt1,
                                           out, states, sync);
}
